// TransformerEmbeddingLM_88759794139587
// MI455X (gfx1250) — matrix operand order repaired and checked by reading, not compiled, not run on hardware
//
#include <hip/hip_runtime.h>
#include <hip/hip_bf16.h>

// ---------------------------------------------------------------------------
// TransformerEmbeddingLM forward for MI455X (gfx1250, wave32, WMMA).
//   EMB=512, NW=64 (S=65 with BOS slot), H=8, dh=64, L=4, FF=2048,
//   VOCAB=50257, BS=16.  Tokens: 1040 (padded to 1056 = 33*32 for GEMM rows).
// All dense GEMMs run on v_wmma_f32_16x16x32_f16 (f16 in, f32 accumulate).
// Wave tile: 32 rows x 64 cols -> 8 WMMAs per 12 b128 loads (B reused 2x).
// ---------------------------------------------------------------------------

#define EMB   512
#define NW    64
#define SEQ   65          // NW + 1
#define HDS   8
#define DH    64
#define NL    4
#define FF    2048
#define VOCAB 50257
#define VPAD  50432       // multiple of 256 (block tile width)
#define BS    16
#define TOK   (BS * SEQ)  // 1040
#define TOKP  1056        // TOK rounded up to multiple of 32
#define OTOK  (BS * NW)   // 1024

typedef __attribute__((ext_vector_type(16))) _Float16 v16h;
typedef __attribute__((ext_vector_type(8)))  _Float16 v8h;
typedef __attribute__((ext_vector_type(8)))  float    v8f;

#define WMMA_F16(a, b, c) \
  __builtin_amdgcn_wmma_f32_16x16x32_f16(false, (a), false, (b), (short)0, (c), false, false)

// ---------------------------------------------------------------------------
// GEMM: C[M,N] = A[M,K] * B[N,K]^T (+bias[n]) (+relu), A/B f16, C f32/f16.
// Block = 128 threads = 4 waves; block tile = 32 rows x 256 cols.
// Each wave: rows [m0, m0+32), cols [n0, n0+64): 2 A fragments, 4 B
// fragments, 8 f32 accumulators -> 8 WMMAs per K-step of 32.
// grid = (N/256, M/32); M multiple of 32, N of 256, K of 32.
//
// A fragment (16x32 f16): lane l -> row m = l&15, contiguous K-run of 16
//   starting at (l>>4)*16 (same k order as B).
// B fragment (32x16 f16): lane l -> col n = l&15, contiguous K-run of 16
//   starting at (l>>4)*16.
// C/D (16x16 f32): lane l -> col n = l&15, vgpr r -> row (l>>4)*8 + r.
// ---------------------------------------------------------------------------
__global__ __launch_bounds__(128)
void gemm_wmma_f16(const _Float16* __restrict__ A, const _Float16* __restrict__ B,
                   const float* __restrict__ bias,
                   float* __restrict__ C, _Float16* __restrict__ C16,
                   int K, int ldc, int nstore, int relu)
{
    const int lane = threadIdx.x & 31;
    const int wave = threadIdx.x >> 5;
    const int m0   = blockIdx.y << 5;
    const int n0   = (blockIdx.x << 8) + (wave << 6);
    const int l15  = lane & 15;
    const int lhi  = lane >> 4;

    const _Float16* ap0 = A + (size_t)(m0 + l15) * K + lhi * 16;
    const _Float16* ap1 = ap0 + (size_t)16 * K;
    const _Float16* bp0 = B + (size_t)(n0 +  0 + l15) * K + lhi * 16;
    const _Float16* bp1 = B + (size_t)(n0 + 16 + l15) * K + lhi * 16;
    const _Float16* bp2 = B + (size_t)(n0 + 32 + l15) * K + lhi * 16;
    const _Float16* bp3 = B + (size_t)(n0 + 48 + l15) * K + lhi * 16;

    v8f a0l = {}, a1l = {}, a2l = {}, a3l = {};   // rows m0..m0+15
    v8f a0h = {}, a1h = {}, a2h = {}, a3h = {};   // rows m0+16..m0+31

    for (int k = 0; k < K; k += 32) {
        // prefetch the B streams ahead (global_prefetch_b8)
        __builtin_prefetch(bp0 + k + 128, 0, 1);
        __builtin_prefetch(bp1 + k + 128, 0, 1);
        __builtin_prefetch(bp2 + k + 128, 0, 1);
        __builtin_prefetch(bp3 + k + 128, 0, 1);

        v16h fa0, fa1, b0, b1, b2, b3;
        ((v8h*)&fa0)[0] = *(const v8h*)(ap0 + k);
        ((v8h*)&fa0)[1] = *(const v8h*)(ap0 + k + 8);
        ((v8h*)&fa1)[0] = *(const v8h*)(ap1 + k);
        ((v8h*)&fa1)[1] = *(const v8h*)(ap1 + k + 8);
        ((v8h*)&b0)[0]  = *(const v8h*)(bp0 + k);
        ((v8h*)&b0)[1]  = *(const v8h*)(bp0 + k + 8);
        ((v8h*)&b1)[0]  = *(const v8h*)(bp1 + k);
        ((v8h*)&b1)[1]  = *(const v8h*)(bp1 + k + 8);
        ((v8h*)&b2)[0]  = *(const v8h*)(bp2 + k);
        ((v8h*)&b2)[1]  = *(const v8h*)(bp2 + k + 8);
        ((v8h*)&b3)[0]  = *(const v8h*)(bp3 + k);
        ((v8h*)&b3)[1]  = *(const v8h*)(bp3 + k + 8);

        a0l = WMMA_F16(fa0, b0, a0l);
        a1l = WMMA_F16(fa0, b1, a1l);
        a2l = WMMA_F16(fa0, b2, a2l);
        a3l = WMMA_F16(fa0, b3, a3l);
        a0h = WMMA_F16(fa1, b0, a0h);
        a1h = WMMA_F16(fa1, b1, a1h);
        a2h = WMMA_F16(fa1, b2, a2h);
        a3h = WMMA_F16(fa1, b3, a3h);
    }

    const float bias0 = bias ? bias[n0 +  0 + l15] : 0.f;
    const float bias1 = bias ? bias[n0 + 16 + l15] : 0.f;
    const float bias2 = bias ? bias[n0 + 32 + l15] : 0.f;
    const float bias3 = bias ? bias[n0 + 48 + l15] : 0.f;
    const int mrA = m0 + lhi * 8;
    const int mrB = m0 + 16 + lhi * 8;

#pragma unroll
    for (int r = 0; r < 8; ++r) {
        const size_t r0 = (size_t)(mrA + r) * (size_t)ldc;
        const size_t r1 = (size_t)(mrB + r) * (size_t)ldc;
        int n;  float v;

        n = n0 + l15;
        v = a0l[r] + bias0; if (relu) v = fmaxf(v, 0.f);
        if (n < nstore) { if (C) C[r0 + n] = v; if (C16) C16[r0 + n] = (_Float16)v; }
        v = a0h[r] + bias0; if (relu) v = fmaxf(v, 0.f);
        if (n < nstore) { if (C) C[r1 + n] = v; if (C16) C16[r1 + n] = (_Float16)v; }

        n += 16;
        v = a1l[r] + bias1; if (relu) v = fmaxf(v, 0.f);
        if (n < nstore) { if (C) C[r0 + n] = v; if (C16) C16[r0 + n] = (_Float16)v; }
        v = a1h[r] + bias1; if (relu) v = fmaxf(v, 0.f);
        if (n < nstore) { if (C) C[r1 + n] = v; if (C16) C16[r1 + n] = (_Float16)v; }

        n += 16;
        v = a2l[r] + bias2; if (relu) v = fmaxf(v, 0.f);
        if (n < nstore) { if (C) C[r0 + n] = v; if (C16) C16[r0 + n] = (_Float16)v; }
        v = a2h[r] + bias2; if (relu) v = fmaxf(v, 0.f);
        if (n < nstore) { if (C) C[r1 + n] = v; if (C16) C16[r1 + n] = (_Float16)v; }

        n += 16;
        v = a3l[r] + bias3; if (relu) v = fmaxf(v, 0.f);
        if (n < nstore) { if (C) C[r0 + n] = v; if (C16) C16[r0 + n] = (_Float16)v; }
        v = a3h[r] + bias3; if (relu) v = fmaxf(v, 0.f);
        if (n < nstore) { if (C) C[r1 + n] = v; if (C16) C16[r1 + n] = (_Float16)v; }
    }
}

// ---------------------------------------------------------------------------
// Attention: one wave32 per (b, h, q); online softmax over the 65 keys.
// qkv rows are [q(512) | k(512) | v(512)]; each lane owns dims {l, l+32}.
// ---------------------------------------------------------------------------
__global__ __launch_bounds__(128)
void attn_kernel(const float* __restrict__ qkv, float* __restrict__ ctx)
{
    const int wid  = (blockIdx.x << 2) + (threadIdx.x >> 5);  // 4 waves/block
    const int lane = threadIdx.x & 31;
    const int q = wid % SEQ;
    const int h = (wid / SEQ) % HDS;
    const int b = wid / (SEQ * HDS);

    const float scale = 0.125f;  // 1/sqrt(64)
    const float* qrow = qkv + (size_t)(b * SEQ + q) * (3 * EMB) + h * DH;
    const float q0 = qrow[lane], q1 = qrow[lane + 32];

    float mx = -1e30f, lsum = 0.f, a0 = 0.f, a1 = 0.f;
    for (int s = 0; s < SEQ; ++s) {
        const float* krow = qkv + (size_t)(b * SEQ + s) * (3 * EMB) + EMB + h * DH;
        float p = q0 * krow[lane] + q1 * krow[lane + 32];
#pragma unroll
        for (int off = 16; off; off >>= 1) p += __shfl_xor(p, off, 32);
        p *= scale;
        const float nm = fmaxf(mx, p);
        const float c  = __expf(mx - nm);
        const float w  = __expf(p - nm);
        const float* vrow = qkv + (size_t)(b * SEQ + s) * (3 * EMB) + 2 * EMB + h * DH;
        a0   = a0 * c + w * vrow[lane];
        a1   = a1 * c + w * vrow[lane + 32];
        lsum = lsum * c + w;
        mx   = nm;
    }
    const float inv = 1.f / lsum;
    float* crow = ctx + (size_t)(b * SEQ + q) * EMB + h * DH;
    crow[lane]      = a0 * inv;
    crow[lane + 32] = a1 * inv;
}

// ---------------------------------------------------------------------------
// Fused residual + LayerNorm; writes f32 x and its f16 shadow.
// One block (256 threads) per token row of 512.
// ---------------------------------------------------------------------------
__global__ __launch_bounds__(256)
void resid_ln(const float* __restrict__ xin, const float* __restrict__ proj,
              const float* __restrict__ g, const float* __restrict__ be,
              float* __restrict__ xout, _Float16* __restrict__ x16out)
{
    __shared__ float sm[16];
    const int tid = threadIdx.x;
    const size_t base = (size_t)blockIdx.x * EMB;

    float v0 = xin[base + tid]       + proj[base + tid];
    float v1 = xin[base + 256 + tid] + proj[base + 256 + tid];
    float s  = v0 + v1;
    float ss = v0 * v0 + v1 * v1;
#pragma unroll
    for (int off = 16; off; off >>= 1) {
        s  += __shfl_xor(s,  off, 32);
        ss += __shfl_xor(ss, off, 32);
    }
    if ((tid & 31) == 0) { sm[tid >> 5] = s; sm[8 + (tid >> 5)] = ss; }
    __syncthreads();
    if (tid == 0) {
        float ts = 0.f, tss = 0.f;
#pragma unroll
        for (int w = 0; w < 8; ++w) { ts += sm[w]; tss += sm[8 + w]; }
        sm[0] = ts; sm[8] = tss;
    }
    __syncthreads();
    const float mean = sm[0] * (1.f / EMB);
    const float var  = sm[8] * (1.f / EMB) - mean * mean;
    const float rstd = rsqrtf(var + 1e-5f);

    const float o0 = (v0 - mean) * rstd * g[tid]       + be[tid];
    const float o1 = (v1 - mean) * rstd * g[256 + tid] + be[256 + tid];
    xout[base + tid]         = o0;
    xout[base + 256 + tid]   = o1;
    x16out[base + tid]       = (_Float16)o0;
    x16out[base + 256 + tid] = (_Float16)o1;
}

// ---------------------------------------------------------------------------
// Small helpers
// ---------------------------------------------------------------------------
__global__ void f32_to_f16(const float* __restrict__ in, _Float16* __restrict__ out, int n)
{
    const int i = blockIdx.x * blockDim.x + threadIdx.x;
    if (i < n) out[i] = (_Float16)in[i];
}

// table (VOCAB x 512) -> f16 padded to VPAD rows (pad rows = 0)
__global__ void table_pad_f16(const float* __restrict__ in, _Float16* __restrict__ out)
{
    const int i = blockIdx.x * blockDim.x + threadIdx.x;
    if (i >= VPAD * EMB) return;
    const int r = i >> 9;
    out[i] = (r < VOCAB) ? (_Float16)in[(size_t)r * EMB + (i & 511)] : (_Float16)0.f;
}

// x = concat(zeros, w_emb) + word_enc ; also f16 shadow
__global__ void embed_kernel(const float* __restrict__ w_emb, const float* __restrict__ word_enc,
                             float* __restrict__ x32, _Float16* __restrict__ x16)
{
    const int i = blockIdx.x * blockDim.x + threadIdx.x;
    if (i >= TOK * EMB) return;
    const int c = i & 511;
    const int t = i >> 9;
    const int s = t % SEQ;
    const int b = t / SEQ;
    float v = word_enc[s * EMB + c];
    if (s > 0) v += w_emb[((size_t)b * NW + (s - 1)) * EMB + c];
    x32[i] = v;
    x16[i] = (_Float16)v;
}

// gather x[:, :-1] rows into a dense f16 (1024 x 512) operand
__global__ void gather_out_f16(const float* __restrict__ x32, _Float16* __restrict__ a16)
{
    const int i = blockIdx.x * blockDim.x + threadIdx.x;
    if (i >= OTOK * EMB) return;
    const int c = i & 511;
    const int t = i >> 9;
    const int n = t & (NW - 1);
    const int b = t >> 6;
    a16[i] = (_Float16)x32[((size_t)(b * SEQ + n)) * EMB + c];
}

// ---------------------------------------------------------------------------
// Host launcher
// ---------------------------------------------------------------------------
extern "C" void kernel_launch(void* const* d_in, const int* in_sizes, int n_in,
                              void* d_out, int out_size, void* d_ws, size_t ws_size,
                              hipStream_t stream)
{
    (void)in_sizes; (void)n_in; (void)out_size; (void)ws_size;

    const float* w_emb    = (const float*)d_in[0];
    const float* table    = (const float*)d_in[1];
    const float* word_enc = (const float*)d_in[2];
    const float* Wqkv     = (const float*)d_in[3];
    const float* bqkv     = (const float*)d_in[4];
    const float* Wo       = (const float*)d_in[5];
    const float* bo       = (const float*)d_in[6];
    const float* W1       = (const float*)d_in[7];
    const float* b1       = (const float*)d_in[8];
    const float* W2       = (const float*)d_in[9];
    const float* b2       = (const float*)d_in[10];
    const float* g1       = (const float*)d_in[11];
    const float* be1      = (const float*)d_in[12];
    const float* g2       = (const float*)d_in[13];
    const float* be2      = (const float*)d_in[14];
    float* out = (float*)d_out;

    // ---- workspace layout (256-B aligned), ~100 MB total ----
    // Activation f16/f32 GEMM buffers are padded to TOKP=1056 rows so every
    // GEMM runs with M % 32 == 0; pad rows carry benign garbage and are never
    // read by attention / layernorm / the final gather.
    char* ws = (char*)d_ws;
    size_t off = 0;
    auto alloc = [&](size_t bytes) -> char* {
        char* p = ws + off;
        off = (off + bytes + 255) & ~(size_t)255;
        return p;
    };
    _Float16* wqkv16 = (_Float16*)alloc((size_t)NL * 3 * EMB * EMB * 2);
    _Float16* wo16   = (_Float16*)alloc((size_t)NL * EMB * EMB * 2);
    _Float16* w116   = (_Float16*)alloc((size_t)NL * FF * EMB * 2);
    _Float16* w216   = (_Float16*)alloc((size_t)NL * EMB * FF * 2);
    _Float16* tab16  = (_Float16*)alloc((size_t)VPAD * EMB * 2);
    float*    x32    = (float*)   alloc((size_t)TOK  * EMB * 4);
    _Float16* x16    = (_Float16*)alloc((size_t)TOKP * EMB * 2);
    float*    qkv32  = (float*)   alloc((size_t)TOKP * 3 * EMB * 4);
    float*    ctx32  = (float*)   alloc((size_t)TOK  * EMB * 4);
    _Float16* ctx16  = (_Float16*)alloc((size_t)TOKP * EMB * 2);
    float*    proj32 = (float*)   alloc((size_t)TOKP * EMB * 4);
    _Float16* h16    = (_Float16*)alloc((size_t)TOKP * FF * 2);
    _Float16* af16   = (_Float16*)alloc((size_t)OTOK * EMB * 2);

    const int CT = 256;
    auto blks = [](size_t n, int t) { return (unsigned)((n + t - 1) / t); };

    // ---- one-time f32 -> f16 weight conversion ----
    f32_to_f16<<<blks((size_t)NL * 3 * EMB * EMB, CT), CT, 0, stream>>>(Wqkv, wqkv16, NL * 3 * EMB * EMB);
    f32_to_f16<<<blks((size_t)NL * EMB * EMB, CT), CT, 0, stream>>>(Wo, wo16, NL * EMB * EMB);
    f32_to_f16<<<blks((size_t)NL * FF * EMB, CT), CT, 0, stream>>>(W1, w116, NL * FF * EMB);
    f32_to_f16<<<blks((size_t)NL * EMB * FF, CT), CT, 0, stream>>>(W2, w216, NL * EMB * FF);
    table_pad_f16<<<blks((size_t)VPAD * EMB, CT), CT, 0, stream>>>(table, tab16);

    // ---- embedding ----
    embed_kernel<<<blks((size_t)TOK * EMB, CT), CT, 0, stream>>>(w_emb, word_enc, x32, x16);

    // ---- transformer layers ----
    for (int l = 0; l < NL; ++l) {
        // qkv = x @ Wqkv^T + bqkv   (1056 x 1536, K=512)
        gemm_wmma_f16<<<dim3((3 * EMB) / 256, TOKP / 32), 128, 0, stream>>>(
            x16, wqkv16 + (size_t)l * 3 * EMB * EMB, bqkv + l * 3 * EMB,
            qkv32, nullptr, EMB, 3 * EMB, 3 * EMB, 0);

        // attention (8320 waves = 2080 blocks of 4)
        attn_kernel<<<(BS * HDS * SEQ) / 4, 128, 0, stream>>>(qkv32, ctx32);
        f32_to_f16<<<blks((size_t)TOK * EMB, CT), CT, 0, stream>>>(ctx32, ctx16, TOK * EMB);

        // proj = ctx @ Wo^T + bo   (1056 x 512, K=512)
        gemm_wmma_f16<<<dim3(EMB / 256, TOKP / 32), 128, 0, stream>>>(
            ctx16, wo16 + (size_t)l * EMB * EMB, bo + l * EMB,
            proj32, nullptr, EMB, EMB, EMB, 0);
        resid_ln<<<TOK, 256, 0, stream>>>(x32, proj32, g1 + l * EMB, be1 + l * EMB, x32, x16);

        // h = relu(x @ W1^T + b1)  (1056 x 2048, K=512) -> f16 directly
        gemm_wmma_f16<<<dim3(FF / 256, TOKP / 32), 128, 0, stream>>>(
            x16, w116 + (size_t)l * FF * EMB, b1 + l * FF,
            nullptr, h16, EMB, FF, FF, 1);

        // proj = h @ W2^T + b2     (1056 x 512, K=2048)
        gemm_wmma_f16<<<dim3(EMB / 256, TOKP / 32), 128, 0, stream>>>(
            h16, w216 + (size_t)l * EMB * FF, b2 + l * EMB,
            proj32, nullptr, FF, EMB, EMB, 0);
        resid_ln<<<TOK, 256, 0, stream>>>(x32, proj32, g2 + l * EMB, be2 + l * EMB, x32, x16);
    }

    // ---- vocab projection: logits = x[:, :-1] @ table^T  (1024 x 50257, K=512)
    gather_out_f16<<<blks((size_t)OTOK * EMB, CT), CT, 0, stream>>>(x32, af16);
    gemm_wmma_f16<<<dim3(VPAD / 256, OTOK / 32), 128, 0, stream>>>(
        af16, tab16, nullptr, out, nullptr, EMB, VOCAB, VOCAB, 0);
}
